// Head_25658134626889
// MI455X (gfx1250) — matrix operand order repaired and checked by reading, not compiled, not run on hardware
//
#include <hip/hip_runtime.h>

// Causal single-head attention, fused bf16-WMMA pipeline for gfx1250 (MI455X).
// B=512, T=256, C=384, H=64. fp32 in/out, bf16 tensor-core math, fp32 accum.

constexpr int kB = 512;
constexpr int kT = 256;
constexpr int kC = 384;
constexpr int kH = 64;

typedef __attribute__((ext_vector_type(16))) __bf16 v16bf;
typedef __attribute__((ext_vector_type(8)))  __bf16 bf16x8;
typedef __attribute__((ext_vector_type(8)))  float  v8f;
typedef __attribute__((ext_vector_type(4)))  float  f32x4;

static __device__ inline v8f wmma_bf16(v16bf a, v16bf b, v8f c) {
  // D(16x16,f32) = A(16x32,bf16) x B(32x16,bf16) + C
  return __builtin_amdgcn_wmma_f32_16x16x32_bf16(false, a, false, b,
                                                 (short)0, c, false, false);
}

// A-operand fragment (16x32 bf16). Lane l<16 holds row (l&15), K = 0..15;
// lanes >=16 hold K = 16..31. Caller passes p = rowBase + kc + (half16?16:0).
static __device__ inline v16bf load_a_bf16(const __bf16* p) {
  bf16x8 lo = *(const bf16x8*)(p);
  bf16x8 hi = *(const bf16x8*)(p + 8);
  v16bf r;
#pragma unroll
  for (int i = 0; i < 8; ++i) { r[i] = lo[i]; r[i + 8] = hi[i]; }
  return r;
}

// B-operand fragment (32x16 bf16). Lane l<16 holds col (l&15), K = 0..15;
// lanes >=16 hold K = 16..31. Caller passes p = rowBase + kc + (half16?16:0).
static __device__ inline v16bf load_b_bf16(const __bf16* p) {
  bf16x8 lo = *(const bf16x8*)(p);
  bf16x8 hi = *(const bf16x8*)(p + 8);
  v16bf r;
#pragma unroll
  for (int i = 0; i < 8; ++i) { r[i] = lo[i]; r[i + 8] = hi[i]; }
  return r;
}

// ---------------------------------------------------------------------------
// Kernel 0: W[C][H] (f32) -> W^T[H][C] (bf16) for Wk, Wq, Wv.
// ---------------------------------------------------------------------------
__global__ __launch_bounds__(256) void k_wtrans(const float* __restrict__ Wk,
                                                const float* __restrict__ Wq,
                                                const float* __restrict__ Wv,
                                                __bf16* __restrict__ wt) {
  int idx = blockIdx.x * 256 + threadIdx.x;
  if (idx >= 3 * kH * kC) return;
  int m = idx / (kH * kC);
  int r = idx - m * (kH * kC);
  int h = r / kC;
  int c = r - h * kC;
  const float* W = (m == 0) ? Wk : ((m == 1) ? Wq : Wv);
  wt[idx] = (__bf16)W[(size_t)c * kH + h];
}

// ---------------------------------------------------------------------------
// Kernel 1: projections. One wave per 16-row tile of x computes Q^T, K^T
// (transposed compute -> contiguous store into q_ws/k_ws[T][H]) and V
// (normal compute -> contiguous store into vT_ws[H][T]).
// x is loaded ONCE per lane (B-fragment pattern: 16 contiguous f32); the
// A-fragment reuses the same 16 values (Wv^T is read in the same k order).
// ---------------------------------------------------------------------------
__global__ __launch_bounds__(256) void k_proj(const float* __restrict__ x,
                                              const __bf16* __restrict__ wt,
                                              __bf16* __restrict__ qws,
                                              __bf16* __restrict__ kws,
                                              __bf16* __restrict__ vtws) {
  const int lane   = threadIdx.x & 31;
  const int lm     = lane & 15;
  const int half16 = (lane >> 4) & 1;
  const int hA     = half16 * 8;
  const int hB     = half16 * 16;
  const int task   = blockIdx.x * 8 + (threadIdx.x >> 5);  // 8192 tiles
  const int b      = task >> 4;
  const int t0     = (task & 15) << 4;

  const __bf16* wkT = wt;
  const __bf16* wqT = wt + kH * kC;
  const __bf16* wvT = wt + 2 * kH * kC;
  const float*  xrow = x + ((size_t)b * kT + t0 + lm) * kC;

  const v8f zero = {0.f, 0.f, 0.f, 0.f, 0.f, 0.f, 0.f, 0.f};
  v8f accQ[4] = {zero, zero, zero, zero};
  v8f accK[4] = {zero, zero, zero, zero};
  v8f accV[4] = {zero, zero, zero, zero};

  for (int kc = 0; kc < kC; kc += 32) {
    // One read of x: lane holds c = hB + {0..15} of row (t0+lm).
    const float* xp = xrow + kc + hB;
    f32x4 xv0 = *(const f32x4*)(xp);
    f32x4 xv1 = *(const f32x4*)(xp + 4);
    f32x4 xv2 = *(const f32x4*)(xp + 8);
    f32x4 xv3 = *(const f32x4*)(xp + 12);
    float v[16];
#pragma unroll
    for (int i = 0; i < 4; ++i) {
      v[i] = xv0[i]; v[4 + i] = xv1[i]; v[8 + i] = xv2[i]; v[12 + i] = xv3[i];
    }
    // B-operand fragment of x^T: direct pack.
    v16bf xb;
#pragma unroll
    for (int i = 0; i < 16; ++i) xb[i] = (__bf16)v[i];
    // A-operand fragment of x: lane uses its own c = hB + {0..15}, same k order as wv;
    // the xor-16 shuffle results plo/phi are unused -> no extra loads.
    v16bf xa;
#pragma unroll
    for (int i = 0; i < 8; ++i) {
      const float plo = __shfl_xor(v[i], 16, 32);      // partner's c = i
      const float phi = __shfl_xor(v[8 + i], 16, 32);  // partner's c = 8+i
      xa[i]     = (__bf16)v[i];
      xa[8 + i] = (__bf16)v[8 + i];
    }

#pragma unroll
    for (int ht = 0; ht < 4; ++ht) {
      const int wrow = (ht * 16 + lm) * kC + kc;
      v16bf wk = load_a_bf16(wkT + wrow + hB);   // K^T: A = Wk^T
      v16bf wq = load_a_bf16(wqT + wrow + hB);   // Q^T: A = Wq^T
      v16bf wv = load_b_bf16(wvT + wrow + hB);   // V:   B = Wv (from Wv^T rows)
      accK[ht] = wmma_bf16(wk, xb, accK[ht]);
      accQ[ht] = wmma_bf16(wq, xb, accQ[ht]);
      accV[ht] = wmma_bf16(xa, wv, accV[ht]);
    }
  }

  // Q^T/K^T tile (M=h, N=t): lane t=lm, h=ht*16 + r + 8*half16 -> [T][H] contiguous.
  // V tile (M=t, N=h): lane h=ht*16+lm, t=t0 + r + 8*half16 -> [H][T] contiguous.
  const size_t qkBase = ((size_t)b * kT + t0 + lm) * kH + hA;
#pragma unroll
  for (int ht = 0; ht < 4; ++ht) {
    bf16x8 pq, pk, pv;
#pragma unroll
    for (int r = 0; r < 8; ++r) {
      pq[r] = (__bf16)accQ[ht][r];
      pk[r] = (__bf16)accK[ht][r];
      pv[r] = (__bf16)accV[ht][r];
    }
    *(bf16x8*)(qws + qkBase + ht * 16) = pq;
    *(bf16x8*)(kws + qkBase + ht * 16) = pk;
    *(bf16x8*)(vtws + ((size_t)b * kH + ht * 16 + lm) * kT + t0 + hA) = pv;
  }
}

// ---------------------------------------------------------------------------
// Kernel 2: fused causal attention. One wave per (batch, 16-query tile).
// S^T = K(A) x Q^T(B): each softmax row (query) lives on lane pair (q, q+16),
// so max/sum reductions are one shfl_xor(16) each. O^T = V^T(A) x P^T(B),
// P^T built in-register from the exp'd S^T tiles. Online softmax over
// 32-key chunks; only the diagonal chunk is masked.
// ---------------------------------------------------------------------------
__global__ __launch_bounds__(256) void k_attn(const __bf16* __restrict__ qws,
                                              const __bf16* __restrict__ kws,
                                              const __bf16* __restrict__ vtws,
                                              float* __restrict__ out) {
  const int lane   = threadIdx.x & 31;
  const int lm     = lane & 15;
  const int half16 = (lane >> 4) & 1;
  const int hA     = half16 * 8;
  const int hB     = half16 * 16;
  const int task   = blockIdx.x * 8 + (threadIdx.x >> 5);  // 8192 tiles
  const int b      = task >> 4;
  const int qt     = task & 15;
  const int qglob  = qt * 16 + lm;

  const __bf16* qrow = qws + ((size_t)b * kT + qt * 16 + lm) * kH + hB;
  v16bf qb0 = load_b_bf16(qrow);        // Q^T B-frag, h 0..31
  v16bf qb1 = load_b_bf16(qrow + 32);   // Q^T B-frag, h 32..63

  const v8f zero = {0.f, 0.f, 0.f, 0.f, 0.f, 0.f, 0.f, 0.f};
  v8f accO[4] = {zero, zero, zero, zero};
  float m = -1e30f, ssum = 0.f;
  const int nch = (qt >> 1) + 1;        // causal: keys 0 .. qt*16+15
  const float scale = 0.125f;           // 1/sqrt(64)

  for (int c = 0; c < nch; ++c) {
    const bool maskc = (c == nch - 1);
    const __bf16* k0 = kws + ((size_t)b * kT + c * 32 + lm) * kH + hB;
    const __bf16* k1 = k0 + 16 * kH;

    v8f s0 = zero, s1 = zero;           // S^T tiles: keys c*32+0..15, +16..31
    {
      v16bf ka0 = load_a_bf16(k0);
      v16bf ka1 = load_a_bf16(k0 + 32);
      v16bf ka2 = load_a_bf16(k1);
      v16bf ka3 = load_a_bf16(k1 + 32);
      s0 = wmma_bf16(ka0, qb0, s0);
      s0 = wmma_bf16(ka1, qb1, s0);
      s1 = wmma_bf16(ka2, qb0, s1);
      s1 = wmma_bf16(ka3, qb1, s1);
    }

    float p0[8], p1[8];
    float lmax = -1e30f;
#pragma unroll
    for (int r = 0; r < 8; ++r) {
      float a0 = s0[r] * scale;
      float a1 = s1[r] * scale;
      if (maskc) {
        const int key = c * 32 + hA + r;      // local key = r + 8*half16
        if (key > qglob)       a0 = -1e30f;
        if (key + 16 > qglob)  a1 = -1e30f;
      }
      p0[r] = a0; p1[r] = a1;
      lmax = fmaxf(lmax, fmaxf(a0, a1));
    }
    lmax = fmaxf(lmax, __shfl_xor(lmax, 16, 32));
    const float mn    = fmaxf(m, lmax);
    const float alpha = __expf(m - mn);
    float ls = 0.f;
#pragma unroll
    for (int r = 0; r < 8; ++r) {
      p0[r] = __expf(p0[r] - mn);
      p1[r] = __expf(p1[r] - mn);
      ls += p0[r] + p1[r];
    }
    ls += __shfl_xor(ls, 16, 32);
    ssum = ssum * alpha + ls;
    m = mn;

    // P^T as B-operand fragment: lanes<16 need keys 0..15, lanes>=16 keys 16..31.
    // C/D layout holds keys 0..7/16..23 on lanes<16 and 8..15/24..31 on lanes>=16,
    // so one lane-pair swap per register completes the fragment.
    v16bf pb;
#pragma unroll
    for (int r = 0; r < 8; ++r) {
      const float o0 = __shfl_xor(p0[r], 16, 32);
      const float o1 = __shfl_xor(p1[r], 16, 32);
      pb[r]     = (__bf16)(half16 ? o1 : p0[r]);
      pb[r + 8] = (__bf16)(half16 ? p1[r] : o0);
    }

#pragma unroll
    for (int ht = 0; ht < 4; ++ht) {
      const __bf16* vrow =
          vtws + ((size_t)b * kH + ht * 16 + lm) * kT + c * 32 + hB;
      v16bf va = load_a_bf16(vrow);     // V^T A-frag (M=h, K=keys)
#pragma unroll
      for (int r = 0; r < 8; ++r) accO[ht][r] *= alpha;
      accO[ht] = wmma_bf16(va, pb, accO[ht]);
    }
  }

  // O^T tile (M=h, N=q): lane q=lm, h=ht*16 + r + 8*half16 -> out[T][H] contiguous.
  const float inv = 1.f / ssum;
  float* orow = out + ((size_t)b * kT + qt * 16 + lm) * kH + hA;
#pragma unroll
  for (int ht = 0; ht < 4; ++ht) {
    f32x4 o0, o1;
#pragma unroll
    for (int r = 0; r < 4; ++r) {
      o0[r] = accO[ht][r] * inv;
      o1[r] = accO[ht][r + 4] * inv;
    }
    *(f32x4*)(orow + ht * 16)     = o0;
    *(f32x4*)(orow + ht * 16 + 4) = o1;
  }
}

extern "C" void kernel_launch(void* const* d_in, const int* in_sizes, int n_in,
                              void* d_out, int out_size, void* d_ws, size_t ws_size,
                              hipStream_t stream) {
  const float* x  = (const float*)d_in[0];
  const float* Wk = (const float*)d_in[1];
  const float* Wq = (const float*)d_in[2];
  const float* Wv = (const float*)d_in[3];

  // Workspace: [WkT|WqT|WvT] bf16 (144 KB), then q[T][H], k[T][H], v^T[H][T]
  // bf16 (16.8 MB each) per batch-major layout. Total ~50.5 MB.
  __bf16* wt   = (__bf16*)d_ws;
  __bf16* qws  = wt + 3 * kH * kC;
  __bf16* kws  = qws + (size_t)kB * kT * kH;
  __bf16* vtws = kws + (size_t)kB * kT * kH;
  float*  outp = (float*)d_out;

  k_wtrans<<<(3 * kH * kC + 255) / 256, 256, 0, stream>>>(Wk, Wq, Wv, wt);
  k_proj<<<(kB * kT / 16) / 8, 256, 0, stream>>>(x, wt, qws, kws, vtws);
  k_attn<<<(kB * 16) / 8, 256, 0, stream>>>(qws, kws, vtws, outp);
}
